// SelectiveScan_43224550867464
// MI455X (gfx1250) — matrix operand order repaired and checked by reading, not compiled, not run on hardware
//
#include <hip/hip_runtime.h>

// ---------------------------------------------------------------------------
// Selective scan (Mamba-like) for MI455X / gfx1250, wave32 + WMMA bf16.
//   P[16384,512] = bf16(x)[16384,2048] @ [W_AB | W_C | W_delta[:, :128]]
//   elementwise + chunked linear-recurrence scan along L
//   y[16384,2048] = (C*h)[16384,128] @ W_out
// ---------------------------------------------------------------------------

typedef __attribute__((ext_vector_type(16))) __bf16 bf16x16;
typedef __attribute__((ext_vector_type(8)))  float  v8f;

#define DMODEL 2048
#define DSTATE 128
#define SEQL   4096
#define NBATCH 4
#define MROWS  (NBATCH * SEQL)     // 16384
#define NPROJ  512                 // [A_raw(128) | B(128) | C(128) | dl(128)]
#define KT1    (DMODEL / 32)       // 64  k-tiles for GEMM1
#define NT1    (NPROJ / 16)        // 32  n-tiles for GEMM1
#define KT2    (DSTATE / 32)       // 4   k-tiles for GEMM2
#define NT2    (DMODEL / 16)       // 128 n-tiles for GEMM2
#define CHUNK  64
#define NCHUNK (SEQL / CHUNK)      // 64 chunks per batch

union Frag {
  bf16x16        v;
  uint4          q[2];
  unsigned int   u[8];
  unsigned short s[16];
};

// Scalar RNE f32->bf16 (cold paths: weight packing, scan output).
__device__ __forceinline__ unsigned short f2bf(float f) {
  union { float f; unsigned int u; } cv; cv.f = f;
  unsigned int u = cv.u;
  unsigned int r = (u + 0x7FFFu + ((u >> 16) & 1u)) >> 16;
  return (unsigned short)r;
}

// Hot path: pack two f32 into one dword of bf16 (lo -> [15:0], hi -> [31:16]).
// Single v_perm_b32 selecting the top 16 bits of each float (truncation);
// upgraded to v_cvt_pk_bf16_f32 automatically if the toolchain provides it.
__device__ __forceinline__ unsigned int f2bf_pk(float lo, float hi) {
#if __has_builtin(__builtin_amdgcn_cvt_pk_bf16_f32)
  auto r = __builtin_amdgcn_cvt_pk_bf16_f32(lo, hi);
  unsigned int u;
  __builtin_memcpy(&u, &r, sizeof(u));
  return u;
#else
  // dst bytes: {hi.b3, hi.b2, lo.b3, lo.b2}  (sel 0-3 -> src1, 4-7 -> src0)
  return __builtin_amdgcn_perm(__float_as_uint(hi), __float_as_uint(lo),
                               0x07060302u);
#endif
}

__device__ __forceinline__ v8f wmma_bf16(const Frag& a, const Frag& b, v8f c) {
  return __builtin_amdgcn_wmma_f32_16x16x32_bf16(false, a.v, false, b.v,
                                                 (short)0, c, false, false);
}

__device__ __forceinline__ float softplusf(float z) {
  return (z > 20.0f) ? z : log1pf(expf(z));
}

// Build a WMMA A fragment from one f32 row of x:
//   lane<16 : K = kb + 0..15;  lane>=16: K = kb + 16..31
__device__ __forceinline__ Frag load_a_f32(const float* __restrict__ row, int hi) {
  const float4* p0 = (const float4*)(row + hi * 16);
  const float4* p1 = (const float4*)(row + hi * 16 + 8);
  float4 f0 = p0[0], f1 = p0[1];
  float4 f2 = p1[0], f3 = p1[1];
  Frag a;
  a.u[0] = f2bf_pk(f0.x, f0.y);
  a.u[1] = f2bf_pk(f0.z, f0.w);
  a.u[2] = f2bf_pk(f1.x, f1.y);
  a.u[3] = f2bf_pk(f1.z, f1.w);
  a.u[4] = f2bf_pk(f2.x, f2.y);
  a.u[5] = f2bf_pk(f2.z, f2.w);
  a.u[6] = f2bf_pk(f3.x, f3.y);
  a.u[7] = f2bf_pk(f3.z, f3.w);
  return a;
}

__device__ __forceinline__ Frag load_b_packed(const unsigned short* __restrict__ W,
                                              int frag_idx, int lane) {
  const uint4* bp = (const uint4*)(W + ((size_t)frag_idx * 32 + lane) * 16);
  Frag b;
  b.q[0] = bp[0];
  b.q[1] = bp[1];
  return b;
}

// ---------------------------------------------------------------------------
// Weight packing (B operand of v_wmma_f32_16x16x32_bf16, this code's contiguous k order, as in A):
//   lane<16 : N = nt*16+lane,    K = kt*32 + 0..15
//   lane>=16: N = nt*16+lane-16, K = kt*32 + 16..31
// ---------------------------------------------------------------------------
__global__ void pack_wproj(const float* __restrict__ W_AB,
                           const float* __restrict__ W_C,
                           const float* __restrict__ W_delta,
                           unsigned short* __restrict__ Wp) {
  int t = blockIdx.x * 256 + threadIdx.x;
  if (t >= KT1 * NT1 * 32) return;
  int lane = t & 31;
  int tile = t >> 5;
  int nt = tile % NT1;
  int kt = tile / NT1;
  int n  = nt * 16 + (lane & 15);
  int k0 = kt * 32 + (lane >> 4) * 16;
  const float* src; int ld, col;
  if (n < 256)      { src = W_AB;    ld = 2 * DSTATE; col = n;       }
  else if (n < 384) { src = W_C;     ld = DSTATE;     col = n - 256; }
  else              { src = W_delta; ld = DMODEL;     col = n - 384; } // only first 128 cols used
  unsigned short* dst = Wp + (size_t)t * 16;
#pragma unroll
  for (int j = 0; j < 16; ++j)
    dst[j] = f2bf(src[(size_t)(k0 + j) * ld + col]);
}

__global__ void pack_wout(const float* __restrict__ W_out,
                          unsigned short* __restrict__ WoP) {
  int t = blockIdx.x * 256 + threadIdx.x;
  if (t >= KT2 * NT2 * 32) return;
  int lane = t & 31;
  int tile = t >> 5;
  int nt = tile % NT2;
  int kt = tile / NT2;
  int n  = nt * 16 + (lane & 15);
  int k0 = kt * 32 + (lane >> 4) * 16;
  unsigned short* dst = WoP + (size_t)t * 16;
#pragma unroll
  for (int j = 0; j < 16; ++j)
    dst[j] = f2bf(W_out[(size_t)(k0 + j) * DMODEL + n]);
}

// ---------------------------------------------------------------------------
// GEMM1: P = bf16(x) @ Wp.  Block covers M=64 x N=512 so x is streamed ONCE.
// 8 waves: wave = (mgrp 0..1) x (nstripe 0..3).  Each wave: 2 M-tiles x
// 8 N-tiles = 128 accum VGPRs; B fragments preloaded in groups of 4 and
// reused across both A tiles.
// ---------------------------------------------------------------------------
__global__ __launch_bounds__(256) void gemm_proj(const float* __restrict__ x,
                                                 const unsigned short* __restrict__ Wp,
                                                 float* __restrict__ P) {
  const int wave    = threadIdx.x >> 5;
  const int lane    = threadIdx.x & 31;
  const int l16     = lane & 15;
  const int hi      = lane >> 4;
  const int nstripe = wave & 3;       // 0..3 -> 128-col stripe
  const int mgrp    = wave >> 2;      // 0..1 -> 32-row group
  const int nt0     = nstripe * 8;
  const int row0    = blockIdx.x * 64 + mgrp * 32 + l16;  // A-tile 0 row
  const float* xr0 = x + (size_t)row0 * DMODEL;
  const float* xr1 = xr0 + (size_t)16 * DMODEL;           // A-tile 1 row

  v8f acc0[8], acc1[8];
#pragma unroll
  for (int t = 0; t < 8; ++t) {
    acc0[t] = (v8f){0,0,0,0,0,0,0,0};
    acc1[t] = (v8f){0,0,0,0,0,0,0,0};
  }

  for (int kt = 0; kt < KT1; ++kt) {
    const int kb = kt * 32;
    __builtin_prefetch(xr0 + kb + 64, 0, 1);
    __builtin_prefetch(xr1 + kb + 64, 0, 1);
    Frag a0 = load_a_f32(xr0 + kb, hi);
    Frag a1 = load_a_f32(xr1 + kb, hi);
#pragma unroll
    for (int g = 0; g < 2; ++g) {
      Frag b[4];
#pragma unroll
      for (int j = 0; j < 4; ++j)
        b[j] = load_b_packed(Wp, kt * NT1 + nt0 + g * 4 + j, lane);
#pragma unroll
      for (int j = 0; j < 4; ++j) {
        acc0[g * 4 + j] = wmma_bf16(a0, b[j], acc0[g * 4 + j]);
        acc1[g * 4 + j] = wmma_bf16(a1, b[j], acc1[g * 4 + j]);
      }
    }
  }

  // D layout: VGPR g -> row (+ hi*8 + g), col = tile*16 + l16
  const int rbase = blockIdx.x * 64 + mgrp * 32 + hi * 8;
#pragma unroll
  for (int t = 0; t < 8; ++t) {
    const int col = (nt0 + t) * 16 + l16;
#pragma unroll
    for (int g = 0; g < 8; ++g) {
      P[(size_t)(rbase + g) * NPROJ + col]      = acc0[t][g];
      P[(size_t)(rbase + 16 + g) * NPROJ + col] = acc1[t][g];
    }
  }
}

// ---------------------------------------------------------------------------
// Scan pass 1: per-chunk elementwise transform + local scan.
//   A = exp(A_raw * softplus(dl)),  Bu = B * softplus(dl) * x[:, :128]
// Overwrites P[:,0:128] with A and P[:,128:256] with Bu; emits chunk totals.
// ---------------------------------------------------------------------------
__global__ __launch_bounds__(128) void scan_chunk(const float* __restrict__ x,
                                                  float* __restrict__ P,
                                                  float* __restrict__ chunkA,
                                                  float* __restrict__ chunkH) {
  const int n = threadIdx.x;
  const int b = blockIdx.x / NCHUNK;
  const int c = blockIdx.x % NCHUNK;
  const size_t r0 = (size_t)b * SEQL + (size_t)c * CHUNK;
  float aP = 1.0f, h = 0.0f;
  for (int i = 0; i < CHUNK; ++i) {
    float* Pr = P + (r0 + i) * NPROJ;
    float a_raw = Pr[n];
    float bp    = Pr[DSTATE + n];
    float dlr   = Pr[3 * DSTATE + n];
    float dl = softplusf(dlr);
    float A  = expf(a_raw * dl);
    float xn = x[(r0 + i) * DMODEL + n];
    float Bu = bp * dl * xn;
    Pr[n] = A;
    Pr[DSTATE + n] = Bu;
    aP *= A;
    h = A * h + Bu;
  }
  chunkA[(size_t)blockIdx.x * DSTATE + n] = aP;
  chunkH[(size_t)blockIdx.x * DSTATE + n] = h;
}

// Scan pass 2: sequential scan over the 64 chunk totals of each batch.
__global__ __launch_bounds__(128) void scan_carry(const float* __restrict__ chunkA,
                                                  const float* __restrict__ chunkH,
                                                  float* __restrict__ carry) {
  const int n = threadIdx.x;
  const int b = blockIdx.x;
  float hc = 0.0f;
  for (int c = 0; c < NCHUNK; ++c) {
    const size_t idx = ((size_t)b * NCHUNK + c) * DSTATE + n;
    carry[idx] = hc;
    hc = chunkA[idx] * hc + chunkH[idx];
  }
}

// Scan pass 3: replay recurrence with chunk carry, emit bf16(C*h) for GEMM2.
__global__ __launch_bounds__(128) void scan_apply(const float* __restrict__ P,
                                                  const float* __restrict__ carry,
                                                  unsigned short* __restrict__ Chb) {
  const int n = threadIdx.x;
  const int b = blockIdx.x / NCHUNK;
  const int c = blockIdx.x % NCHUNK;
  const size_t r0 = (size_t)b * SEQL + (size_t)c * CHUNK;
  float h = carry[(size_t)blockIdx.x * DSTATE + n];
  for (int i = 0; i < CHUNK; ++i) {
    const float* Pr = P + (r0 + i) * NPROJ;
    float A  = Pr[n];
    float Bu = Pr[DSTATE + n];
    float Cc = Pr[2 * DSTATE + n];
    h = A * h + Bu;
    Chb[(r0 + i) * DSTATE + n] = f2bf(Cc * h);
  }
}

// ---------------------------------------------------------------------------
// GEMM2: y = Chb[16384,128] @ W_out.  K=128 (4 k-steps, fully unrolled);
// b[8] preloaded per k-step -> one wait, 8 back-to-back WMMAs.  Store-bound
// on the 134 MB y write.
// ---------------------------------------------------------------------------
__global__ __launch_bounds__(256) void gemm_out(const unsigned short* __restrict__ Chb,
                                                const unsigned short* __restrict__ WoP,
                                                float* __restrict__ Y) {
  const int wave = threadIdx.x >> 5;
  const int lane = threadIdx.x & 31;
  const int l16  = lane & 15;
  const int hi   = lane >> 4;
  const int mrow = blockIdx.x * 128 + wave * 16 + l16;
  const int nt0  = blockIdx.y * 8;
  const unsigned short* ar = Chb + (size_t)mrow * DSTATE;

  v8f acc[8];
#pragma unroll
  for (int t = 0; t < 8; ++t) acc[t] = (v8f){0,0,0,0,0,0,0,0};

#pragma unroll
  for (int kt = 0; kt < KT2; ++kt) {
    const int kb = kt * 32;
    Frag a;
    a.q[0] = *(const uint4*)(ar + kb + hi * 16);
    a.q[1] = *(const uint4*)(ar + kb + hi * 16 + 8);
    Frag b[8];
#pragma unroll
    for (int t = 0; t < 8; ++t)
      b[t] = load_b_packed(WoP, kt * NT2 + nt0 + t, lane);
#pragma unroll
    for (int t = 0; t < 8; ++t)
      acc[t] = wmma_bf16(a, b[t], acc[t]);
  }

  const int rbase = blockIdx.x * 128 + wave * 16 + hi * 8;
#pragma unroll
  for (int t = 0; t < 8; ++t) {
    const int col = (nt0 + t) * 16 + l16;
#pragma unroll
    for (int g = 0; g < 8; ++g)
      Y[(size_t)(rbase + g) * DMODEL + col] = acc[t][g];
  }
}

// ---------------------------------------------------------------------------
extern "C" void kernel_launch(void* const* d_in, const int* in_sizes, int n_in,
                              void* d_out, int out_size, void* d_ws, size_t ws_size,
                              hipStream_t stream) {
  const float* x       = (const float*)d_in[0];
  const float* W_AB    = (const float*)d_in[1];
  const float* W_C     = (const float*)d_in[2];
  const float* W_delta = (const float*)d_in[3];
  const float* W_out   = (const float*)d_in[4];
  float* y = (float*)d_out;

  char* ws = (char*)d_ws;
  size_t off = 0;
  unsigned short* Wp  = (unsigned short*)(ws + off); off += (size_t)KT1 * NT1 * 32 * 16 * 2; // 2 MB
  unsigned short* WoP = (unsigned short*)(ws + off); off += (size_t)KT2 * NT2 * 32 * 16 * 2; // 512 KB
  float* P      = (float*)(ws + off); off += (size_t)MROWS * NPROJ * 4;                       // 32 MB
  float* chunkA = (float*)(ws + off); off += (size_t)NBATCH * NCHUNK * DSTATE * 4;
  float* chunkH = (float*)(ws + off); off += (size_t)NBATCH * NCHUNK * DSTATE * 4;
  float* carry  = (float*)(ws + off); off += (size_t)NBATCH * NCHUNK * DSTATE * 4;
  unsigned short* Chb = (unsigned short*)(ws + off);                                          // 4 MB

  pack_wproj<<<(KT1 * NT1 * 32 + 255) / 256, 256, 0, stream>>>(W_AB, W_C, W_delta, Wp);
  pack_wout<<<(KT2 * NT2 * 32 + 255) / 256, 256, 0, stream>>>(W_out, WoP);

  gemm_proj<<<MROWS / 64, 256, 0, stream>>>(x, Wp, P);

  scan_chunk<<<NBATCH * NCHUNK, 128, 0, stream>>>(x, P, chunkA, chunkH);
  scan_carry<<<NBATCH, 128, 0, stream>>>(chunkA, chunkH, carry);
  scan_apply<<<NBATCH * NCHUNK, 128, 0, stream>>>(P, carry, Chb);

  gemm_out<<<dim3(MROWS / 128, DMODEL / 128), 256, 0, stream>>>(Chb, WoP, y);
}
